// NeuroKernel_91182155694480
// MI455X (gfx1250) — matrix operand order repaired and checked by reading, not compiled, not run on hardware
//
#include <hip/hip_runtime.h>
#include <hip/hip_bf16.h>
#include <math.h>
#include <stdint.h>

typedef __attribute__((ext_vector_type(16))) _Float16 v16h;
typedef __attribute__((ext_vector_type(8)))  float    v8f;
typedef __attribute__((ext_vector_type(2)))  float    v2f;

#define NN 1024

// ---------------------------------------------------------------------------
// Kernel 1: build K (upper-triangular, unit diagonal) with the pair MLP.
// One wave handles one 16-entry row segment: pairs (x[i], x[j0+m]), m=0..15.
// Layer1 (2->128 + tanh): VALU/TRANS, results staged to LDS as f16.
// Layer2 (128->32 + relu): v_wmma_f32_16x16x32_f16, 8 WMMAs per tile.
// Layer3 (32->1): small dot product via LDS.
// Only wave-local s_wait_dscnt syncs are used, so below-diagonal tiles can
// early-exit without deadlocking a block barrier.
// ---------------------------------------------------------------------------
__global__ __launch_bounds__(256) void build_k_kernel(
    const float* __restrict__ x,
    const float* __restrict__ W1, const float* __restrict__ b1,
    const float* __restrict__ W2, const float* __restrict__ b2,
    const float* __restrict__ W3, const float* __restrict__ b3,
    float* __restrict__ K)
{
    __shared__ _Float16 ldsH [8][16 * 128];  // per-wave hidden tile (4 KB each)
    __shared__ float    ldsH2[8][16 * 32];   // per-wave layer2 output (2 KB each)

    const int lane = threadIdx.x & 31;
    const int wave = threadIdx.x >> 5;
    const int tile = blockIdx.x * 8 + wave;  // 1024 * 64 = 65536 tiles
    const int i    = tile >> 6;
    const int j0   = (tile & 63) << 4;

    // Entire segment strictly below the diagonal -> zeros, no MLP.
    if (j0 + 15 < i) {
        if (lane < 16) K[i * NN + j0 + lane] = 0.0f;
        return;
    }

    const float xi = x[i];

    // ---- Layer 1: lane computes 64 features for pair p = lane%16 ----------
    {
        const int p  = lane & 15;
        const int f0 = (lane >> 4) * 64;
        const float xj = x[j0 + p];
        for (int f = f0; f < f0 + 64; ++f) {
            float h = tanhf(xi * W1[2 * f + 0] + xj * W1[2 * f + 1] + b1[f]);
            ldsH[wave][p * 128 + f] = (_Float16)h;
        }
    }
    asm volatile("s_wait_dscnt 0x0" ::: "memory");  // wave-local LDS RAW fence

    // ---- Preload B fragments of W2^T (constant for all tiles) -------------
    // B fragment, k order chosen here (same as A): lane -> column n = lane%16;
    // lanes 0-15 hold K=0..15, lanes 16-31 hold K=16..31 (contiguous in k).
    v16h Bf[2][4];
    {
        const int n    = lane & 15;
        const int koff = (lane < 16) ? 0 : 16;
        for (int nt = 0; nt < 2; ++nt)
            for (int kt = 0; kt < 4; ++kt) {
                const float* src = W2 + (nt * 16 + n) * 128 + kt * 32 + koff;
                v16h bfrag;
                for (int e = 0; e < 16; ++e) bfrag[e] = (_Float16)src[e];
                Bf[nt][kt] = bfrag;
            }
    }

    // ---- Layer 2: H[16x128] @ W2^T[128x32] via 8 WMMAs --------------------
    v8f acc0 = {}, acc1 = {};
    for (int kt = 0; kt < 4; ++kt) {
        // A fragment, k order chosen here (same as B): lane -> row m = lane%16;
        // lanes 0-15: K = 0..15; lanes 16-31: K = 16..31 (contiguous in k).
        const int m     = lane & 15;
        const int kbase = kt * 32 + ((lane < 16) ? 0 : 16);
        const _Float16* hp = &ldsH[wave][m * 128 + kbase];
        v16h a;
        for (int e = 0; e < 8; ++e) a[e]     = hp[e];
        for (int e = 0; e < 8; ++e) a[8 + e] = hp[8 + e];
        acc0 = __builtin_amdgcn_wmma_f32_16x16x32_f16(
                   false, a, false, Bf[0][kt], (short)0, acc0, false, false);
        acc1 = __builtin_amdgcn_wmma_f32_16x16x32_f16(
                   false, a, false, Bf[1][kt], (short)0, acc1, false, false);
    }

    // ---- bias + relu, scatter to LDS in plain [pair][neuron] order --------
    // C/D layout: lane -> col n = lane%16; VGPR r -> row m = r + (lane<16?0:8)
    {
        const int n     = lane & 15;
        const int mbase = (lane < 16) ? 0 : 8;
        const float bb0 = b2[n], bb1 = b2[16 + n];
        for (int r = 0; r < 8; ++r) {
            float v0 = acc0[r] + bb0; v0 = v0 > 0.0f ? v0 : 0.0f;
            float v1 = acc1[r] + bb1; v1 = v1 > 0.0f ? v1 : 0.0f;
            ldsH2[wave][(mbase + r) * 32 + n]      = v0;
            ldsH2[wave][(mbase + r) * 32 + 16 + n] = v1;
        }
    }
    asm volatile("s_wait_dscnt 0x0" ::: "memory");

    // ---- Layer 3 (32->1) + masked store into K ----------------------------
    if (lane < 16) {
        const int j = j0 + lane;
        float v = b3[0];
        for (int n = 0; n < 32; ++n) v += ldsH2[wave][lane * 32 + n] * W3[n];
        float out = (j > i) ? v : ((j == i) ? 1.0f : 0.0f);
        K[i * NN + j] = out;
    }
}

// ---------------------------------------------------------------------------
// Kernel 2: C = sigma^2 * K^T K  via v_wmma_f32_16x16x4_f32.
// Block (256 threads, 8 waves) computes a 64x64 C tile; wave w owns a
// 32x16 strip (2 accumulators). K-panels of 32 are staged into LDS with the
// CDNA5 async global->LDS path (GLOBAL_LOAD_ASYNC_TO_LDS_B128, ASYNCcnt),
// bypassing VGPRs entirely; completion via s_wait_asynccnt + block barrier.
// ---------------------------------------------------------------------------
__global__ __launch_bounds__(256) void ktk_kernel(
    const float* __restrict__ K,
    const float* __restrict__ sigma,
    float* __restrict__ C)
{
    __shared__ float ldsA[32 * 64];  // K[kk..kk+31][a0..a0+63]
    __shared__ float ldsB[32 * 64];  // K[kk..kk+31][b0..b0+63]

    const int lane = threadIdx.x & 31;
    const int wave = threadIdx.x >> 5;
    const int a0   = blockIdx.y * 64;
    const int b0   = blockIdx.x * 64;
    const int wr   = wave >> 2;      // 0..1 : 32-row strip
    const int wc   = wave & 3;       // 0..3 : 16-col strip

    const int nloc = wc * 16 + (lane & 15);   // local B/C column
    const int koff = (lane < 16) ? 0 : 2;     // f32 A/B: lanes 0-15 K={0,1}, 16-31 K={2,3}

    // LDS byte offsets (flat shared address low 32 bits == wave LDS offset).
    const uint32_t ldsA_base = (uint32_t)(uintptr_t)(void*)ldsA;
    const uint32_t ldsB_base = (uint32_t)(uintptr_t)(void*)ldsB;

    v8f acc[2] = {v8f{}, v8f{}};

    for (int kk = 0; kk < NN; kk += 32) {
        __syncthreads();   // previous panel fully consumed before overwrite

        // ---- async panel stage: each thread moves 2x16B per panel ---------
        // chunk c covers floats [c*4 .. c*4+3] of the 32x64 panel.
        #pragma unroll
        for (int rep = 0; rep < 2; ++rep) {
            const int chunk = threadIdx.x + rep * 256;   // 0..511
            const int r     = chunk >> 4;                // panel row 0..31
            const int col   = (chunk & 15) * 4;          // panel col 0..60
            const uint64_t gA = (uint64_t)(uintptr_t)(K + (kk + r) * NN + a0 + col);
            const uint64_t gB = (uint64_t)(uintptr_t)(K + (kk + r) * NN + b0 + col);
            const uint32_t lA = ldsA_base + (uint32_t)chunk * 16u;
            const uint32_t lB = ldsB_base + (uint32_t)chunk * 16u;
            asm volatile("global_load_async_to_lds_b128 %0, %1, off"
                         :: "v"(lA), "v"(gA) : "memory");
            asm volatile("global_load_async_to_lds_b128 %0, %1, off"
                         :: "v"(lB), "v"(gB) : "memory");
        }
        asm volatile("s_wait_asynccnt 0x0" ::: "memory");
        __syncthreads();

        // ---- compute: 8 K-substeps of 4, 2 WMMAs each ---------------------
        for (int ks = 0; ks < 32; ks += 4) {
            // B fragment: lane -> column n, K = ks+koff, ks+koff+1
            v2f bf = { ldsB[(ks + koff) * 64 + nloc],
                       ldsB[(ks + koff + 1) * 64 + nloc] };
            #pragma unroll
            for (int s = 0; s < 2; ++s) {
                const int mloc = wr * 32 + s * 16 + (lane & 15);
                v2f af = { ldsA[(ks + koff) * 64 + mloc],
                           ldsA[(ks + koff + 1) * 64 + mloc] };
                acc[s] = __builtin_amdgcn_wmma_f32_16x16x4_f32(
                             false, af, false, bf, (short)0, acc[s], false, false);
            }
        }
    }

    // Scale by sigma^2 and store. C/D layout: lane -> col, VGPR r -> row.
    const float s2  = sigma[0] * sigma[0];
    const int   col = b0 + nloc;
    for (int s = 0; s < 2; ++s) {
        const int mbase = a0 + wr * 32 + s * 16 + ((lane < 16) ? 0 : 8);
        for (int r = 0; r < 8; ++r)
            C[(mbase + r) * NN + col] = acc[s][r] * s2;
    }
}

// ---------------------------------------------------------------------------
extern "C" void kernel_launch(void* const* d_in, const int* in_sizes, int n_in,
                              void* d_out, int out_size, void* d_ws, size_t ws_size,
                              hipStream_t stream) {
    const float* x     = (const float*)d_in[0];
    const float* W1    = (const float*)d_in[1];
    const float* b1    = (const float*)d_in[2];
    const float* W2    = (const float*)d_in[3];
    const float* b2    = (const float*)d_in[4];
    const float* W3    = (const float*)d_in[5];
    const float* b3    = (const float*)d_in[6];
    const float* sigma = (const float*)d_in[7];

    float* Kmat = (float*)d_ws;          // 1024*1024*4 = 4 MB scratch
    float* Cout = (float*)d_out;

    // Phase 1: build K (65536 row-segment tiles, 8 waves per block)
    build_k_kernel<<<dim3(65536 / 8), dim3(256), 0, stream>>>(
        x, W1, b1, W2, b2, W3, b3, Kmat);

    // Phase 2: C = sigma^2 * K^T K  (16x16 grid of 64x64 tiles)
    ktk_kernel<<<dim3(16, 16), dim3(256), 0, stream>>>(Kmat, sigma, Cout);
}
